// UE_Mamba_decoder_Gated_50706383897027
// MI455X (gfx1250) — matrix operand order repaired and checked by reading, not compiled, not run on hardware
//
#include <hip/hip_runtime.h>
#include <cstdint>
#include <cstddef>

// ---------------------------------------------------------------------------
// Problem constants (from the reference)
// ---------------------------------------------------------------------------
constexpr int B_   = 2;
constexpr int C_   = 128;
constexpr int H_   = 64;
constexpr int W_   = 64;
constexpr int L_   = H_ * W_;          // 4096
constexpr int M_   = B_ * L_;          // 8192 tokens (multiple of 16)
constexpr int NST  = 16;               // SSM state size
constexpr int Ch_  = C_ / 2;           // 64
constexpr int COUT = 128;

// ---------------------------------------------------------------------------
// WMMA vector types (gfx1250, wave32)
// ---------------------------------------------------------------------------
typedef __attribute__((ext_vector_type(16))) _Float16 v16h;
typedef __attribute__((ext_vector_type(8)))  _Float16 v8h;
typedef __attribute__((ext_vector_type(8)))  float    v8f;

// ---------------------------------------------------------------------------
// Branch-free GEMM:  Co[M, Npad] = A[M, >=KR](f16) * Wp[Npad, KR](f16,packed)
//                                 + biasp[Npad]
// One wave per 16x16 tile.  KR is a compile-time constant so the K loop fully
// unrolls into straight-line global_load_b128 -> v_wmma_f32_16x16x32_f16
// chains.  All raggedness (N=36/40, K=4/8) was removed by zero-padded packing,
// so there are no guards anywhere: EXEC stays all-ones (WMMA requirement) and
// every access is an aligned 128-bit load / 32-bit store.
// ---------------------------------------------------------------------------
template <int KR>
__global__ __launch_bounds__(32)
void wmma_gemm(const _Float16* __restrict__ A, int lda,
               const _Float16* __restrict__ Wp,
               const float* __restrict__ biasp,
               float* __restrict__ Co, int ldc, int ntile)
{
    const int lane = threadIdx.x & 31;
    const int half = lane >> 4;          // half-wave id
    const int lrow = lane & 15;
    const int mt   = blockIdx.x / ntile;
    const int nt   = blockIdx.x % ntile;

    const _Float16* __restrict__ arow = A  + (long)(mt * 16 + lrow) * lda;
    const _Float16* __restrict__ brow = Wp + (long)(nt * 16 + lrow) * KR;

    v8f acc = {};
#pragma unroll
    for (int k0 = 0; k0 < KR; k0 += 32) {
        // A fragment: element i -> k = k0 + half*8 + (i<8 ? i : i+8)
        v8h a0 = *(const v8h*)(arow + k0 + half * 8);
        v8h a1 = *(const v8h*)(arow + k0 + half * 8 + 16);
        // B fragment: element i -> k = k0 + half*8 + (i<8 ? i : i+8)
        v8h b0 = *(const v8h*)(brow + k0 + half * 8);
        v8h b1 = *(const v8h*)(brow + k0 + half * 8 + 16);
        v16h af, bf;
#pragma unroll
        for (int j = 0; j < 8; ++j) {
            af[j] = a0[j]; af[8 + j] = a1[j];
            bf[j] = b0[j]; bf[8 + j] = b1[j];
        }
        acc = __builtin_amdgcn_wmma_f32_16x16x32_f16(
                  false, af, false, bf, (short)0, acc, false, false);
    }

    const float bv = biasp[nt * 16 + lrow];
    float* __restrict__ ccol = Co + (long)(mt * 16 + half * 8) * ldc
                                  + nt * 16 + lrow;
#pragma unroll
    for (int r = 0; r < 8; ++r)
        ccol[(long)r * ldc] = acc[r] + bv;
}

// ---------------------------------------------------------------------------
// Weight / bias packing: f32 [Nr, Kr] -> zero-padded f16 [Np, Kp]
// (tiny, hoists the f32->f16 conversion out of the GEMM inner loop)
// ---------------------------------------------------------------------------
__global__ void pack_w_f16(const float* __restrict__ Wsrc, int Nr, int Kr,
                           _Float16* __restrict__ out, int Np, int Kp)
{
    long i = (long)blockIdx.x * blockDim.x + threadIdx.x;
    if (i >= (long)Np * Kp) return;
    const int n = (int)(i / Kp);
    const int k = (int)(i % Kp);
    out[i] = (n < Nr && k < Kr) ? (_Float16)Wsrc[(long)n * Kr + k]
                                : (_Float16)0.f;
}

__global__ void pack_bias(const float* __restrict__ b, int Nr,
                          float* __restrict__ out, int Np)
{
    int i = blockIdx.x * blockDim.x + threadIdx.x;
    if (i < Np) out[i] = (i < Nr) ? b[i] : 0.f;
}

// A-side pack: f32 [rows, ldin] cols [0,cols) -> zero-padded f16 [rows, ldo]
__global__ void pack_a_f16(const float* __restrict__ in, int ldin, int cols,
                           _Float16* __restrict__ out, int ldo, long rows)
{
    long i = (long)blockIdx.x * blockDim.x + threadIdx.x;
    if (i >= rows * ldo) return;
    const long r = i / ldo;
    const int  c = (int)(i % ldo);
    out[i] = (c < cols) ? (_Float16)in[r * ldin + c] : (_Float16)0.f;
}

// ---------------------------------------------------------------------------
// Channel LayerNorm (eps 1e-6).  One wave per token; NCHW or token-major.
// ---------------------------------------------------------------------------
__global__ __launch_bounds__(32)
void ln_cf(const float* __restrict__ in, int ldi,
           const float* __restrict__ w, const float* __restrict__ b,
           float* __restrict__ out, int ldo,
           int Cn, int in_nchw, int out_nchw)
{
    const int r    = blockIdx.x;
    const int lane = threadIdx.x;
    const int bb   = r / L_;
    const int hw   = r % L_;

    long ibase, istr, obase, ostr;
    if (in_nchw)  { ibase = (long)bb * Cn * L_ + hw; istr = L_; }
    else          { ibase = (long)r * ldi;           istr = 1;  }
    if (out_nchw) { obase = (long)bb * Cn * L_ + hw; ostr = L_; }
    else          { obase = (long)r * ldo;           ostr = 1;  }

    float s = 0.f, s2 = 0.f;
    for (int c = lane; c < Cn; c += 32) {
        float v = in[ibase + (long)c * istr];
        s += v; s2 += v * v;
    }
#pragma unroll
    for (int off = 16; off > 0; off >>= 1) {
        s  += __shfl_xor(s,  off, 32);
        s2 += __shfl_xor(s2, off, 32);
    }
    const float mu  = s / Cn;
    const float var = s2 / Cn - mu * mu;
    const float rs  = rsqrtf(var + 1e-6f);
    for (int c = lane; c < Cn; c += 32)
        out[obase + (long)c * ostr] =
            w[c] * (in[ibase + (long)c * istr] - mu) * rs + b[c];
}

// ---------------------------------------------------------------------------
// Elementwise helpers
// ---------------------------------------------------------------------------
__global__ void f32_to_f16(const float* __restrict__ in,
                           _Float16* __restrict__ out, long n)
{
    long i = (long)blockIdx.x * blockDim.x + threadIdx.x;
    if (i < n) out[i] = (_Float16)in[i];
}

__global__ void concat2_f16(const float* __restrict__ a, int lda,
                            const float* __restrict__ b2, int ldb,
                            _Float16* __restrict__ out, int Ca, int Cb)
{
    const int Ct = Ca + Cb;
    long i = (long)blockIdx.x * blockDim.x + threadIdx.x;
    if (i >= (long)M_ * Ct) return;
    const long r = i / Ct;
    const int  c = (int)(i % Ct);
    const float v = (c < Ca) ? a[r * lda + c] : b2[r * ldb + (c - Ca)];
    out[i] = (_Float16)v;
}

__global__ void softplus_ip(float* __restrict__ x, long n)
{
    long i = (long)blockIdx.x * blockDim.x + threadIdx.x;
    if (i < n) {
        float v = x[i];
        x[i] = (v > 20.f) ? v : log1pf(expf(v));
    }
}

__global__ void bn_relu_ip(float* __restrict__ g,
                           const float* __restrict__ gm,
                           const float* __restrict__ be,
                           const float* __restrict__ mn,
                           const float* __restrict__ vr,
                           int Cn, long n)
{
    long i = (long)blockIdx.x * blockDim.x + threadIdx.x;
    if (i >= n) return;
    const int c = (int)(i % Cn);
    float v = (g[i] - mn[c]) * rsqrtf(vr[c] + 1e-5f);
    v = v * gm[c] + be[c];
    g[i] = v > 0.f ? v : 0.f;
}

__global__ void gate_fuse(const float* __restrict__ glog,
                          const float* __restrict__ sid,
                          const float* __restrict__ cat,
                          float* __restrict__ fused, long n)
{
    long i = (long)blockIdx.x * blockDim.x + threadIdx.x;
    if (i >= n) return;
    const float g = 1.f / (1.f + expf(-glog[i]));
    fused[i] = g * sid[i] + (1.f - g) * cat[i];
}

// ---------------------------------------------------------------------------
// Depthwise 3x3 conv (SAME) + SiLU, token-major layouts.
// ---------------------------------------------------------------------------
__global__ void dwconv_silu(const float* __restrict__ in, int ldin, int coff,
                            const float* __restrict__ wk,
                            const float* __restrict__ bias,
                            float* __restrict__ out, int ldo, int Cn)
{
    long i = (long)blockIdx.x * blockDim.x + threadIdx.x;
    if (i >= (long)M_ * Cn) return;
    const int  c   = (int)(i % Cn);
    const long pos = i / Cn;
    const int  bb  = (int)(pos / L_);
    const int  hw  = (int)(pos % L_);
    const int  h   = hw >> 6;
    const int  w   = hw & 63;

    float acc = bias[c];
#pragma unroll
    for (int dy = 0; dy < 3; ++dy) {
        const int hh = h + dy - 1;
        if (hh < 0 || hh >= H_) continue;
#pragma unroll
        for (int dx = 0; dx < 3; ++dx) {
            const int ww = w + dx - 1;
            if (ww < 0 || ww >= W_) continue;
            const long p = (long)bb * L_ + hh * W_ + ww;
            acc += in[p * ldin + coff + c] * wk[c * 9 + dy * 3 + dx];
        }
    }
    out[pos * ldo + c] = acc / (1.f + expf(-acc));   // SiLU
}

// ---------------------------------------------------------------------------
// Priority order: argsort(-(cat + l*1e-4)) == buckets cat 3..0, within a
// bucket descending l (stable ascending sort of negated key).
// ---------------------------------------------------------------------------
__global__ void prio_sort(const float* __restrict__ fg,
                          const float* __restrict__ bg,
                          const float* __restrict__ uc,
                          int* __restrict__ order)
{
    const int bb = blockIdx.x;
    if (threadIdx.x != 0) return;
    int idx = 0;
    for (int cat = 3; cat >= 0; --cat) {
        for (int l = L_ - 1; l >= 0; --l) {
            const long p = (long)bb * L_ + l;
            const int cc = (fg[p] > 0.5f) ? 3
                         : (bg[p] > 0.5f) ? 2
                         : (uc[p] > 0.5f) ? 1 : 0;
            if (cc == cat) order[(long)bb * L_ + (idx++)] = l;
        }
    }
}

// ---------------------------------------------------------------------------
// Selective scan.  One wave per (batch, 2 channels); lane = (pair half,
// state n).  Reversal / priority permutation are pure index remaps because
// the per-position projections commute with position permutations, and
// writing y at the processed position cancels the inverse-argsort scatter.
// ---------------------------------------------------------------------------
__global__ __launch_bounds__(32)
void ssm_scan(const float* __restrict__ delta, int lddt,
              const float* __restrict__ u, int ldu, int ucol,
              const float* __restrict__ dbl, int ldb, int bsoff, int csoff,
              const float* __restrict__ A_log, const float* __restrict__ Dp,
              const int* __restrict__ order, int reverse,
              float* __restrict__ y, int ldy, int ycol, int Cch)
{
    const int wid  = blockIdx.x;
    const int bb   = wid / (Cch >> 1);
    const int cp   = wid % (Cch >> 1);
    const int lane = threadIdx.x;
    const int c    = cp * 2 + (lane >> 4);
    const int n    = lane & 15;

    const float a = -__expf(A_log[c * NST + n]);
    const float d = Dp[c];
    float h = 0.f;

    for (int t = 0; t < L_; ++t) {
        const int p = order ? order[(long)bb * L_ + t]
                            : (reverse ? (L_ - 1 - t) : t);
        const long row = (long)bb * L_ + p;
        const float dl = delta[row * lddt + c];
        const float uu = u[row * ldu + ucol + c];
        const float bs = dbl[row * ldb + bsoff + n];
        const float cs = dbl[row * ldb + csoff + n];

        h = __expf(dl * a) * h + dl * uu * bs;

        float v = h * cs;
        v += __shfl_xor(v, 8, 32);
        v += __shfl_xor(v, 4, 32);
        v += __shfl_xor(v, 2, 32);
        v += __shfl_xor(v, 1, 32);
        if (n == 0) y[row * ldy + ycol + c] = v + uu * d;
    }
}

// ---------------------------------------------------------------------------
// Host-side orchestration
// ---------------------------------------------------------------------------
static inline dim3 ew_grid(long n) { return dim3((unsigned)((n + 255) / 256)); }

extern "C" void kernel_launch(void* const* d_in, const int* in_sizes, int n_in,
                              void* d_out, int out_size, void* d_ws, size_t ws_size,
                              hipStream_t stream)
{
    (void)in_sizes; (void)n_in; (void)out_size; (void)ws_size;

    const float* x         = (const float*)d_in[0];
    const float* mask_fg   = (const float*)d_in[1];
    const float* mask_bg   = (const float*)d_in[2];
    const float* mask_uc   = (const float*)d_in[3];
    const float* norm_in_w = (const float*)d_in[4];
    const float* norm_in_b = (const float*)d_in[5];
    const float* in_proj_w = (const float*)d_in[6];
    const float* in_proj_b = (const float*)d_in[7];
    const float* ln_mam_w  = (const float*)d_in[8];
    const float* ln_mam_b  = (const float*)d_in[9];
    const float* cvm_w     = (const float*)d_in[10];
    const float* cvm_b     = (const float*)d_in[11];
    const float* cvr_w     = (const float*)d_in[12];
    const float* cvr_b     = (const float*)d_in[13];
    const float* xpf_w     = (const float*)d_in[14];
    const float* xpf_b     = (const float*)d_in[15];
    const float* dtf_w     = (const float*)d_in[16];
    const float* dtf_b     = (const float*)d_in[17];
    const float* Alog_f    = (const float*)d_in[18];
    const float* D_f       = (const float*)d_in[19];
    const float* xpb_w     = (const float*)d_in[20];
    const float* xpb_b     = (const float*)d_in[21];
    const float* dtb_w     = (const float*)d_in[22];
    const float* dtb_b     = (const float*)d_in[23];
    const float* Alog_b    = (const float*)d_in[24];
    const float* D_b       = (const float*)d_in[25];
    const float* xps_w     = (const float*)d_in[26];
    const float* xps_b     = (const float*)d_in[27];
    const float* dts_w     = (const float*)d_in[28];
    const float* dts_b     = (const float*)d_in[29];
    const float* Alog_s    = (const float*)d_in[30];
    const float* D_s       = (const float*)d_in[31];
    const float* lncat_w   = (const float*)d_in[32];
    const float* lncat_b   = (const float*)d_in[33];
    const float* g0_w      = (const float*)d_in[34];
    const float* g0_b      = (const float*)d_in[35];
    const float* bn_g      = (const float*)d_in[36];
    const float* bn_be     = (const float*)d_in[37];
    const float* bn_m      = (const float*)d_in[38];
    const float* bn_v      = (const float*)d_in[39];
    const float* g1_w      = (const float*)d_in[40];
    const float* g1_b      = (const float*)d_in[41];
    const float* out_w     = (const float*)d_in[42];
    const float* out_b     = (const float*)d_in[43];
    const float* no_w      = (const float*)d_in[44];
    const float* no_b      = (const float*)d_in[45];
    float* out = (float*)d_out;

    // ---- workspace carve (~80 MB; L2-resident on MI455X) ----
    char* wp = (char*)d_ws;
    auto carve = [&](size_t bytes) -> void* {
        void* p = (void*)wp;
        wp += (bytes + 255) & ~(size_t)255;
        return p;
    };
    // activations
    float*    xln     = (float*)   carve((size_t)M_ * 128 * 4);
    _Float16* xln_h   = (_Float16*)carve((size_t)M_ * 128 * 2);
    float*    xp      = (float*)   carve((size_t)M_ * 256 * 4);
    float*    xm_ln   = (float*)   carve((size_t)M_ * 128 * 4);
    float*    xm_post = (float*)   carve((size_t)M_ * 128 * 4);
    float*    xr_post = (float*)   carve((size_t)M_ * 128 * 4);
    _Float16* xm_h    = (_Float16*)carve((size_t)M_ * 128 * 2);
    float*    dbl_f   = (float*)   carve((size_t)M_ * 48 * 4);
    float*    dbl_b   = (float*)   carve((size_t)M_ * 48 * 4);
    float*    dbl_s   = (float*)   carve((size_t)M_ * 48 * 4);
    _Float16* dtfin_h = (_Float16*)carve((size_t)M_ * 32 * 2);
    _Float16* dtbin_h = (_Float16*)carve((size_t)M_ * 32 * 2);
    _Float16* dtsin_h = (_Float16*)carve((size_t)M_ * 32 * 2);
    float*    delta_f = (float*)   carve((size_t)M_ * 64 * 4);
    float*    delta_b = (float*)   carve((size_t)M_ * 64 * 4);
    float*    delta_s = (float*)   carve((size_t)M_ * 128 * 4);
    float*    ycat    = (float*)   carve((size_t)M_ * 128 * 4);
    float*    ysid    = (float*)   carve((size_t)M_ * 128 * 4);
    float*    cat_ln  = (float*)   carve((size_t)M_ * 128 * 4);
    _Float16* g0in_h  = (_Float16*)carve((size_t)M_ * 256 * 2);
    float*    gbuf    = (float*)   carve((size_t)M_ * 128 * 4);
    _Float16* g_h     = (_Float16*)carve((size_t)M_ * 128 * 2);
    float*    glog    = (float*)   carve((size_t)M_ * 128 * 4);
    float*    fused   = (float*)   carve((size_t)M_ * 128 * 4);
    _Float16* oin_h   = (_Float16*)carve((size_t)M_ * 256 * 2);
    float*    obuf    = (float*)   carve((size_t)M_ * 128 * 4);
    int*      order   = (int*)     carve((size_t)B_ * L_ * 4);
    // packed weights (f16 [Npad, Kpad], zero-filled) and padded biases
    _Float16* w_inpj  = (_Float16*)carve((size_t)256 * 128 * 2);
    _Float16* w_xpf   = (_Float16*)carve((size_t)48 * 64 * 2);
    _Float16* w_xpb   = (_Float16*)carve((size_t)48 * 64 * 2);
    _Float16* w_xps   = (_Float16*)carve((size_t)48 * 128 * 2);
    _Float16* w_dtf   = (_Float16*)carve((size_t)64 * 32 * 2);
    _Float16* w_dtb   = (_Float16*)carve((size_t)64 * 32 * 2);
    _Float16* w_dts   = (_Float16*)carve((size_t)128 * 32 * 2);
    _Float16* w_g0    = (_Float16*)carve((size_t)128 * 256 * 2);
    _Float16* w_g1    = (_Float16*)carve((size_t)128 * 128 * 2);
    _Float16* w_out   = (_Float16*)carve((size_t)128 * 256 * 2);
    float*    b_xpf   = (float*)   carve(48 * 4);
    float*    b_xpb   = (float*)   carve(48 * 4);
    float*    b_xps   = (float*)   carve(48 * 4);

    // ---- 0. pack all GEMM weights/biases (tiny, removes ragged guards) ----
    pack_w_f16<<<ew_grid(256 * 128), 256, 0, stream>>>(in_proj_w, 256, 128, w_inpj, 256, 128);
    pack_w_f16<<<ew_grid(48 * 64),   256, 0, stream>>>(xpf_w, 36, 64,   w_xpf, 48, 64);
    pack_w_f16<<<ew_grid(48 * 64),   256, 0, stream>>>(xpb_w, 36, 64,   w_xpb, 48, 64);
    pack_w_f16<<<ew_grid(48 * 128),  256, 0, stream>>>(xps_w, 40, 128,  w_xps, 48, 128);
    pack_w_f16<<<ew_grid(64 * 32),   256, 0, stream>>>(dtf_w, 64, 4,    w_dtf, 64, 32);
    pack_w_f16<<<ew_grid(64 * 32),   256, 0, stream>>>(dtb_w, 64, 4,    w_dtb, 64, 32);
    pack_w_f16<<<ew_grid(128 * 32),  256, 0, stream>>>(dts_w, 128, 8,   w_dts, 128, 32);
    pack_w_f16<<<ew_grid(128 * 256), 256, 0, stream>>>(g0_w, 128, 256,  w_g0, 128, 256);
    pack_w_f16<<<ew_grid(128 * 128), 256, 0, stream>>>(g1_w, 128, 128,  w_g1, 128, 128);
    pack_w_f16<<<ew_grid(128 * 256), 256, 0, stream>>>(out_w, 128, 256, w_out, 128, 256);
    pack_bias<<<1, 64, 0, stream>>>(xpf_b, 36, b_xpf, 48);
    pack_bias<<<1, 64, 0, stream>>>(xpb_b, 36, b_xpb, 48);
    pack_bias<<<1, 64, 0, stream>>>(xps_b, 40, b_xps, 48);

    // ---- 1. norm_in (NCHW -> token-major) + in_proj GEMM ----
    ln_cf<<<M_, 32, 0, stream>>>(x, 0, norm_in_w, norm_in_b, xln, 128, 128, 1, 0);
    f32_to_f16<<<ew_grid((long)M_ * 128), 256, 0, stream>>>(xln, xln_h, (long)M_ * 128);
    wmma_gemm<128><<<(M_ / 16) * 16, 32, 0, stream>>>(xln_h, 128, w_inpj, in_proj_b, xp, 256, 16);

    // ---- 2. ln_mamba + dwconv/SiLU ----
    ln_cf<<<M_, 32, 0, stream>>>(xp, 256, ln_mam_w, ln_mam_b, xm_ln, 128, 128, 0, 0);
    dwconv_silu<<<ew_grid((long)M_ * 128), 256, 0, stream>>>(xm_ln, 128, 0, cvm_w, cvm_b, xm_post, 128, 128);
    dwconv_silu<<<ew_grid((long)M_ * 128), 256, 0, stream>>>(xp, 256, 128, cvr_w, cvr_b, xr_post, 128, 128);
    f32_to_f16<<<ew_grid((long)M_ * 128), 256, 0, stream>>>(xm_post, xm_h, (long)M_ * 128);

    // ---- 3. x-projections (dt|B|C), padded N=48 ----
    wmma_gemm<64> <<<(M_ / 16) * 3, 32, 0, stream>>>(xm_h,      128, w_xpf, b_xpf, dbl_f, 48, 3);
    wmma_gemm<64> <<<(M_ / 16) * 3, 32, 0, stream>>>(xm_h + 64, 128, w_xpb, b_xpb, dbl_b, 48, 3);
    wmma_gemm<128><<<(M_ / 16) * 3, 32, 0, stream>>>(xm_h,      128, w_xps, b_xps, dbl_s, 48, 3);

    // ---- 4. dt-projections (K padded 4/8 -> 32 via zero-filled A pack) ----
    pack_a_f16<<<ew_grid((long)M_ * 32), 256, 0, stream>>>(dbl_f, 48, 4, dtfin_h, 32, M_);
    pack_a_f16<<<ew_grid((long)M_ * 32), 256, 0, stream>>>(dbl_b, 48, 4, dtbin_h, 32, M_);
    pack_a_f16<<<ew_grid((long)M_ * 32), 256, 0, stream>>>(dbl_s, 48, 8, dtsin_h, 32, M_);
    wmma_gemm<32><<<(M_ / 16) * 4, 32, 0, stream>>>(dtfin_h, 32, w_dtf, dtf_b, delta_f, 64, 4);
    wmma_gemm<32><<<(M_ / 16) * 4, 32, 0, stream>>>(dtbin_h, 32, w_dtb, dtb_b, delta_b, 64, 4);
    wmma_gemm<32><<<(M_ / 16) * 8, 32, 0, stream>>>(dtsin_h, 32, w_dts, dts_b, delta_s, 128, 8);
    softplus_ip<<<ew_grid((long)M_ * 64),  256, 0, stream>>>(delta_f, (long)M_ * 64);
    softplus_ip<<<ew_grid((long)M_ * 64),  256, 0, stream>>>(delta_b, (long)M_ * 64);
    softplus_ip<<<ew_grid((long)M_ * 128), 256, 0, stream>>>(delta_s, (long)M_ * 128);

    // ---- 5. scans: forward, backward (reversed), sid (permuted) ----
    ssm_scan<<<B_ * (Ch_ / 2), 32, 0, stream>>>(delta_f, 64, xm_post, 128, 0,  dbl_f, 48, 4, 20,
                                                Alog_f, D_f, nullptr, 0, ycat, 128, 0,  Ch_);
    ssm_scan<<<B_ * (Ch_ / 2), 32, 0, stream>>>(delta_b, 64, xm_post, 128, 64, dbl_b, 48, 4, 20,
                                                Alog_b, D_b, nullptr, 1, ycat, 128, 64, Ch_);
    prio_sort<<<B_, 32, 0, stream>>>(mask_fg, mask_bg, mask_uc, order);
    ssm_scan<<<B_ * (C_ / 2), 32, 0, stream>>>(delta_s, 128, xm_post, 128, 0, dbl_s, 48, 8, 24,
                                               Alog_s, D_s, order, 0, ysid, 128, 0, C_);

    // ---- 6. ln_cat + gating path ----
    ln_cf<<<M_, 32, 0, stream>>>(ycat, 128, lncat_w, lncat_b, cat_ln, 128, 128, 0, 0);
    concat2_f16<<<ew_grid((long)M_ * 256), 256, 0, stream>>>(cat_ln, 128, ysid, 128, g0in_h, 128, 128);
    wmma_gemm<256><<<(M_ / 16) * 8, 32, 0, stream>>>(g0in_h, 256, w_g0, g0_b, gbuf, 128, 8);
    bn_relu_ip<<<ew_grid((long)M_ * 128), 256, 0, stream>>>(gbuf, bn_g, bn_be, bn_m, bn_v, 128, (long)M_ * 128);
    f32_to_f16<<<ew_grid((long)M_ * 128), 256, 0, stream>>>(gbuf, g_h, (long)M_ * 128);
    wmma_gemm<128><<<(M_ / 16) * 8, 32, 0, stream>>>(g_h, 128, w_g1, g1_b, glog, 128, 8);
    gate_fuse<<<ew_grid((long)M_ * 128), 256, 0, stream>>>(glog, ysid, cat_ln, fused, (long)M_ * 128);

    // ---- 7. out projection + norm_out (token-major -> NCHW) ----
    concat2_f16<<<ew_grid((long)M_ * 256), 256, 0, stream>>>(fused, 128, xr_post, 128, oin_h, 128, 128);
    wmma_gemm<256><<<(M_ / 16) * (COUT / 16), 32, 0, stream>>>(oin_h, 256, w_out, out_b, obuf, 128, COUT / 16);
    ln_cf<<<M_, 32, 0, stream>>>(obuf, 128, no_w, no_b, out, 128, COUT, 0, 1);
}
